// BinarySEMVectorQuantizer_27573690040945
// MI455X (gfx1250) — matrix operand order repaired and checked by reading, not compiled, not run on hardware
//
#include <hip/hip_runtime.h>
#include <hip/hip_bf16.h>

typedef __attribute__((ext_vector_type(16))) _Float16 v16h;
typedef __attribute__((ext_vector_type(8)))  _Float16 v8h;
typedef __attribute__((ext_vector_type(8)))  float    v8f;
typedef __attribute__((ext_vector_type(4)))  int      v4i;

#define IMG      384
#define NPIX     147456      /* 384*384  rows of zf per scale           */
#define NELEM    4718592     /* 32*384*384 elements per tensor          */
#define VOCAB    2048
#define DIMC     32
#define LOSS_BLOCKS 18432    /* NELEM/256                               */

// ---------------------------------------------------------------- async global->LDS (CDNA5), with safe fallbacks
#if __has_builtin(__builtin_amdgcn_global_load_async_to_lds_b128)
#define HAVE_ASYNC_LDS 1
#else
#define HAVE_ASYNC_LDS 0
#endif

__device__ __forceinline__ void cp16_g2l(const _Float16* g, _Float16* l) {
#if HAVE_ASYNC_LDS
    __builtin_amdgcn_global_load_async_to_lds_b128(
        (__attribute__((address_space(1))) v4i*)(void*)g,
        (__attribute__((address_space(3))) v4i*)(void*)l,
        0, 0);
#else
    *(v8h*)l = *(const v8h*)g;
#endif
}
__device__ __forceinline__ void wait_async_lds() {
#if __has_builtin(__builtin_amdgcn_s_wait_asynccnt)
    __builtin_amdgcn_s_wait_asynccnt(0);
#elif HAVE_ASYNC_LDS
    asm volatile("s_wait_asynccnt 0" ::: "memory");
#endif
}

// ---------------------------------------------------------------- utils
__device__ __forceinline__ float blocksum256(float v) {
    #pragma unroll
    for (int off = 16; off; off >>= 1) v += __shfl_xor(v, off, 32);
    __shared__ float sh[8];
    if ((threadIdx.x & 31) == 0) sh[threadIdx.x >> 5] = v;
    __syncthreads();
    float t = 0.f;
    if (threadIdx.x == 0) {
        #pragma unroll
        for (int i = 0; i < 8; i++) t += sh[i];
    }
    return t;   // valid on thread 0 only
}

// ---------------------------------------------------------------- zero acc
__global__ __launch_bounds__(256) void k_zero(float* __restrict__ acc) {
    acc[blockIdx.x * 256 + threadIdx.x] = 0.f;
}

// ---------------------------------------------------------------- pack weights into WMMA A-fragment order (f16)
__global__ __launch_bounds__(256) void k_pack_w(const float* __restrict__ pw1,
                                                const float* __restrict__ pw2,
                                                const float* __restrict__ rw1,
                                                const float* __restrict__ rw2,
                                                _Float16* __restrict__ wpack) {
    int tid  = blockIdx.x * 256 + threadIdx.x;      // 147456 exact
    int c    = tid & 31;
    int o    = (tid >> 5) & 31;
    int t    = (tid >> 10) % 9;
    int conv = tid / (9 * 1024);                    // src*4 + scale, 0..15
    int src  = conv >> 2, scale = conv & 3;
    const float* W = (src == 0) ? pw1 : (src == 1) ? pw2 : (src == 2) ? rw1 : rw2;
    float v = W[(((scale * 32 + o) * 32 + c) * 9) + t];
    int tile = o >> 4;
    int lane = (o & 15) + ((c >> 4) << 4);          // K in {16..31} -> high half-wave
    int i    = c & 15;
    wpack[((conv * 9 + t) * 2 + tile) * 512 + lane * 16 + i] = (_Float16)v;
}

// ---------------------------------------------------------------- codebook: f32 -> f16, and ||e||^2
__global__ __launch_bounds__(256) void k_pack_emb(const float* __restrict__ emb,
                                                  _Float16* __restrict__ embh) {
    int e = blockIdx.x * 256 + threadIdx.x;         // 65536 exact
    embh[e] = (_Float16)emb[e];
}
__global__ __launch_bounds__(256) void k_emb_norm(const float* __restrict__ emb,
                                                  float* __restrict__ e2) {
    int v = blockIdx.x * 256 + threadIdx.x;         // 2048 exact
    const float* r = emb + v * DIMC;
    float s = 0.f;
    #pragma unroll
    for (int k = 0; k < DIMC; k++) s += r[k] * r[k];
    e2[v] = s;
}

// ---------------------------------------------------------------- conv3x3 32->32 implicit GEMM (WMMA f16 -> f32)
// MODE 0: input = x - acc in global (1,32,384,384) layout ; MODE 1: patch layout (Np,32,P,P)
template<int P, int MODE>
__global__ __launch_bounds__(256) void k_conv3x3(const float* __restrict__ in,
                                                 const float* __restrict__ acc,
                                                 const _Float16* __restrict__ wp,
                                                 const float* __restrict__ bias,
                                                 float* __restrict__ out) {
    constexpr int NW  = IMG / P;
    constexpr int PP  = P * P;
    constexpr int GPP = PP >> 4;                    // 16-pixel groups per patch
    int wave  = (blockIdx.x * 256 + threadIdx.x) >> 5;   // 9216 waves exact
    int lane  = threadIdx.x & 31;
    int patch = wave / GPP;
    int g     = wave - patch * GPP;
    int i0 = (g * 16) / P;                          // one row segment (P % 16 == 0)
    int j0 = (g * 16) - i0 * P;
    int ph = patch / NW, pw = patch - ph * NW;
    int n    = lane & 15;
    int half = lane >> 4;
    int chbase = half * 16;
    v8f c0 = {}, c1 = {};
    __builtin_prefetch(wp + lane * 16, 0, 1);
    #pragma unroll
    for (int t = 0; t < 9; t++) {
        int di = t / 3 - 1, dj = t % 3 - 1;
        int si = i0 + di;
        int sj = j0 + n + dj;
        v16h b = {};
        if (si >= 0 && si < P && sj >= 0 && sj < P) {
            if (MODE == 0) {
                int off = (ph * P + si) * IMG + pw * P + sj + chbase * NPIX;
                #pragma unroll
                for (int j = 0; j < 16; j++)
                    b[j] = (_Float16)(in[off + j * NPIX] - acc[off + j * NPIX]);
            } else {
                int off = (patch * 32 + chbase) * PP + si * P + sj;
                #pragma unroll
                for (int j = 0; j < 16; j++)
                    b[j] = (_Float16)in[off + j * PP];
            }
        }
        v16h a0 = *(const v16h*)(wp + ((t * 2 + 0) * 32 + lane) * 16);
        v16h a1 = *(const v16h*)(wp + ((t * 2 + 1) * 32 + lane) * 16);
        c0 = __builtin_amdgcn_wmma_f32_16x16x32_f16(false, a0, false, b, (short)0, c0, false, false);
        c1 = __builtin_amdgcn_wmma_f32_16x16x32_f16(false, a1, false, b, (short)0, c1, false, false);
    }
    // D layout: lane l, vgpr r -> M = r + 8*half (+16 per tile), pixel N = n
    int mbase = half * 8;
    int pix   = i0 * P + j0 + n;
    int obase = patch * 32 * PP + pix;
    #pragma unroll
    for (int r = 0; r < 8; r++) {
        int m0 = r + mbase, m1 = m0 + 16;
        out[obase + m0 * PP] = c0[r] + bias[m0];
        out[obase + m1 * PP] = c1[r] + bias[m1];
    }
}

// ---------------------------------------------------------------- GroupNorm(8) + exact GELU, in place
template<int P>
__global__ __launch_bounds__(256) void k_gn_gelu(float* __restrict__ h,
                                                 const float* __restrict__ gamma,
                                                 const float* __restrict__ beta) {
    constexpr int PP = P * P;
    constexpr int NE = 4 * PP;
    int patch = blockIdx.x >> 3;
    int grp   = blockIdx.x & 7;
    float* base = h + (patch * 32 + grp * 4) * PP;
    float s = 0.f, ss = 0.f;
    for (int e = threadIdx.x; e < NE; e += 256) { float v = base[e]; s += v; ss += v * v; }
    #pragma unroll
    for (int off = 16; off; off >>= 1) { s += __shfl_xor(s, off, 32); ss += __shfl_xor(ss, off, 32); }
    __shared__ float sh_s[8], sh_ss[8], sh_mu, sh_rstd;
    if ((threadIdx.x & 31) == 0) { sh_s[threadIdx.x >> 5] = s; sh_ss[threadIdx.x >> 5] = ss; }
    __syncthreads();
    if (threadIdx.x == 0) {
        float S = 0.f, SS = 0.f;
        #pragma unroll
        for (int i = 0; i < 8; i++) { S += sh_s[i]; SS += sh_ss[i]; }
        float mu = S / NE;
        sh_mu = mu;
        sh_rstd = rsqrtf(SS / NE - mu * mu + 1e-5f);
    }
    __syncthreads();
    float mu = sh_mu, rstd = sh_rstd;
    for (int e = threadIdx.x; e < NE; e += 256) {
        int ch = grp * 4 + e / PP;
        float v = (base[e] - mu) * rstd * gamma[ch] + beta[ch];
        base[e] = 0.5f * v * (1.0f + erff(v * 0.70710678118f));
    }
}

// ---------------------------------------------------------------- VQ argmin via WMMA, codebook + ||e||^2 staged in LDS
// score = z.e - 0.5||e||^2 (C operand carries the -0.5||e||^2 bias);
// argmin dist == argmax score. Double-buffered async staging, 16 stages
// x 128 codes (8KB codebook f16 + 512B e2 f32), shared by 8 waves.
__global__ __launch_bounds__(256) void k_vq(const float* __restrict__ z,
                                            const _Float16* __restrict__ embh,
                                            const float* __restrict__ e2,
                                            int* __restrict__ idx) {
    __shared__ _Float16 lbuf[2][128 * DIMC];        // 2 x 8KB
    __shared__ float    le2[2][128];                // 2 x 512B
    int tid  = threadIdx.x;
    int w    = tid >> 5;
    int lane = tid & 31;
    int m    = lane & 15;
    int half = lane >> 4;
    int row0 = (blockIdx.x * 8 + w) * 16;

    // A fragment: 16 zf rows, K=32 (f32 -> f16)
    const float* zr = z + (row0 + m) * DIMC;
    v16h a;
    #pragma unroll
    for (int i = 0; i < 16; i++) {
        int k = half * 16 + i;
        a[i] = (_Float16)zr[k];
    }
    float best[8]; int bidx[8];
    #pragma unroll
    for (int r = 0; r < 8; r++) { best[r] = -3.4e38f; bidx[r] = 0; }

    // prologue: stage 0 -> buffer 0
    {
        #pragma unroll
        for (int q = 0; q < 2; q++) {
            int off = (tid * 2 + q) * 8;
            cp16_g2l(embh + off, &lbuf[0][off]);
        }
        if (tid < 32)
            cp16_g2l((const _Float16*)e2 + tid * 8, (_Float16*)&le2[0][0] + tid * 8);
    }
    for (int st = 0; st < VOCAB / 128; st++) {
        wait_async_lds();
        __syncthreads();                            // stage st resident for all waves
        if (st + 1 < VOCAB / 128) {                 // overlap stage st+1 with compute
            const _Float16* gs = embh + (st + 1) * 128 * DIMC;
            _Float16* ls = lbuf[(st + 1) & 1];
            #pragma unroll
            for (int q = 0; q < 2; q++) {
                int off = (tid * 2 + q) * 8;
                cp16_g2l(gs + off, ls + off);
            }
            if (tid < 32)
                cp16_g2l((const _Float16*)(e2 + (st + 1) * 128) + tid * 8,
                         (_Float16*)&le2[(st + 1) & 1][0] + tid * 8);
        }
        const _Float16* lb = lbuf[st & 1];
        const float*    l2 = le2[st & 1];
        // software pipeline: B/e2 for chunk cc+1 loaded between WMMA(cc) and its use
        v16h  b  = *(const v16h*)(lb + m * DIMC + half * 16);
        float ec = l2[m];
        #pragma unroll
        for (int cc = 0; cc < 8; cc++) {
            v8f c;
            float ch = -0.5f * ec;
            #pragma unroll
            for (int r = 0; r < 8; r++) c[r] = ch;
            v8f d = __builtin_amdgcn_wmma_f32_16x16x32_f16(false, a, false, b, (short)0, c, false, false);
            if (cc < 7) {
                b  = *(const v16h*)(lb + ((cc + 1) * 16 + m) * DIMC + half * 16);
                ec = l2[(cc + 1) * 16 + m];
            }
            int code = st * 128 + cc * 16 + m;
            #pragma unroll
            for (int r = 0; r < 8; r++) {
                if (d[r] > best[r]) { best[r] = d[r]; bidx[r] = code; }
            }
        }
        __syncthreads();                            // all waves done with buf[st&1]
    }
    // max-reduce across the 16 lanes of each half; tie -> lowest index
    #pragma unroll
    for (int off = 1; off < 16; off <<= 1) {
        #pragma unroll
        for (int r = 0; r < 8; r++) {
            float ov = __shfl_xor(best[r], off, 32);
            int   oi = __shfl_xor(bidx[r], off, 32);
            if (ov > best[r] || (ov == best[r] && oi < bidx[r])) { best[r] = ov; bidx[r] = oi; }
        }
    }
    if (m == 0) {
        #pragma unroll
        for (int r = 0; r < 8; r++) idx[row0 + half * 8 + r] = bidx[r];
    }
}

// ---------------------------------------------------------------- gather zq = emb[idx] + partial SSE (deterministic)
__global__ __launch_bounds__(256) void k_gather_loss(const float* __restrict__ z,
                                                     const float* __restrict__ emb,
                                                     const int* __restrict__ idx,
                                                     float* __restrict__ zq,
                                                     float* __restrict__ part) {
    int e = blockIdx.x * 256 + threadIdx.x;         // NELEM exact
    int row = e >> 5, c = e & 31;
    float q  = emb[idx[row] * DIMC + c];
    float zv = z[e];
    zq[e] = q;
    float d = q - zv;
    float t = blocksum256(d * d);
    if (threadIdx.x == 0) part[blockIdx.x] = t;
}

// ---------------------------------------------------------------- 0.5*zq + 0.5*r, fold into acc (global layout)
template<int P>
__global__ __launch_bounds__(256) void k_combine_fold(const float* __restrict__ zq,
                                                      const float* __restrict__ r,
                                                      float* __restrict__ acc) {
    constexpr int PP = P * P;
    constexpr int NW = IMG / P;
    int e = blockIdx.x * 256 + threadIdx.x;         // NELEM exact
    float v = 0.5f * zq[e] + 0.5f * r[e];
    int patch = e / (32 * PP);
    int rem   = e - patch * 32 * PP;
    int c   = rem / PP;
    int pix = rem - c * PP;
    int i = pix / P, j = pix - i * P;
    int ph = patch / NW, pw = patch - ph * NW;
    acc[(c * IMG + ph * P + i) * IMG + pw * P + j] += v;
}

// ---------------------------------------------------------------- epilogues
__global__ __launch_bounds__(256) void k_sigmoid(const float* __restrict__ acc,
                                                 float* __restrict__ out) {
    int e = blockIdx.x * 256 + threadIdx.x;
    out[e] = 1.0f / (1.0f + expf(-acc[e]));
}
__global__ __launch_bounds__(256) void k_idx_out(const int* __restrict__ idx,
                                                 float* __restrict__ out) {
    int e = blockIdx.x * 256 + threadIdx.x;         // 589824 exact
    out[e] = (float)idx[e];
}
__global__ __launch_bounds__(256) void k_loss_out(const float* __restrict__ part,
                                                  float* __restrict__ out) {
    float v = 0.f;
    for (int i = threadIdx.x; i < 4 * LOSS_BLOCKS; i += 256) v += part[i];
    float t = blocksum256(v);
    if (threadIdx.x == 0) out[0] = t * (1.25f / (float)NELEM);
}

// ================================================================ per-scale schedule (P compile-time)
struct Args {
    const float *x, *emb;
    const float *pb1, *pg, *pbt, *pb2, *rb1, *rg, *rbt, *rb2;
    float *acc, *zbuf, *zq, *bufA;
    _Float16 *embh, *wpack;
    float *e2;
    int *idxb;
    float *part;
};

template<int P>
static void launch_scale(const Args& A, int s, hipStream_t stream) {
    constexpr int NW = IMG / P;
    constexpr int NPATCH = NW * NW;
    const int convBlocks = (NPIX / 16) * 32 / 256;   // 1152, EXEC all-ones
    const _Float16* wPhi1 = A.wpack + (size_t)(0 * 4 + s) * 9216;
    const _Float16* wPhi2 = A.wpack + (size_t)(1 * 4 + s) * 9216;
    const _Float16* wRes1 = A.wpack + (size_t)(2 * 4 + s) * 9216;
    const _Float16* wRes2 = A.wpack + (size_t)(3 * 4 + s) * 9216;

    // phi conv block on (x - acc) patches
    k_conv3x3<P, 0><<<convBlocks, 256, 0, stream>>>(A.x, A.acc, wPhi1, A.pb1 + s * 32, A.bufA);
    k_gn_gelu<P><<<NPATCH * 8, 256, 0, stream>>>(A.bufA, A.pg + s * 32, A.pbt + s * 32);
    k_conv3x3<P, 1><<<convBlocks, 256, 0, stream>>>(A.bufA, A.acc, wPhi2, A.pb2 + s * 32, A.zbuf);

    // vector quantize + gather + loss partials
    k_vq<<<convBlocks, 256, 0, stream>>>(A.zbuf, A.embh, A.e2, A.idxb + (size_t)s * NPIX);
    k_gather_loss<<<NELEM / 256, 256, 0, stream>>>(A.zbuf, A.emb, A.idxb + (size_t)s * NPIX,
                                                   A.zq, A.part + (size_t)s * LOSS_BLOCKS);

    // residual conv block on zq (zbuf free after loss -> reuse for r)
    k_conv3x3<P, 1><<<convBlocks, 256, 0, stream>>>(A.zq, A.acc, wRes1, A.rb1 + s * 32, A.bufA);
    k_gn_gelu<P><<<NPATCH * 8, 256, 0, stream>>>(A.bufA, A.rg + s * 32, A.rbt + s * 32);
    k_conv3x3<P, 1><<<convBlocks, 256, 0, stream>>>(A.bufA, A.acc, wRes2, A.rb2 + s * 32, A.zbuf);

    // 0.5*zq + 0.5*r folded into acc
    k_combine_fold<P><<<NELEM / 256, 256, 0, stream>>>(A.zq, A.zbuf, A.acc);
}

// ================================================================ launch
extern "C" void kernel_launch(void* const* d_in, const int* in_sizes, int n_in,
                              void* d_out, int out_size, void* d_ws, size_t ws_size,
                              hipStream_t stream) {
    const float* x   = (const float*)d_in[0];
    const float* emb = (const float*)d_in[1];
    const float* pw1 = (const float*)d_in[2];
    const float* pb1 = (const float*)d_in[3];
    const float* pg  = (const float*)d_in[4];
    const float* pbt = (const float*)d_in[5];
    const float* pw2 = (const float*)d_in[6];
    const float* pb2 = (const float*)d_in[7];
    const float* rw1 = (const float*)d_in[8];
    const float* rb1 = (const float*)d_in[9];
    const float* rg  = (const float*)d_in[10];
    const float* rbt = (const float*)d_in[11];
    const float* rw2 = (const float*)d_in[12];
    const float* rb2 = (const float*)d_in[13];
    float* out = (float*)d_out;

    // ---- workspace carve-up (~78.6 MB) -------------------------------
    char* ws = (char*)d_ws;
    Args A;
    A.x = x; A.emb = emb;
    A.pb1 = pb1; A.pg = pg; A.pbt = pbt; A.pb2 = pb2;
    A.rb1 = rb1; A.rg = rg; A.rbt = rbt; A.rb2 = rb2;
    A.acc   = (float*)(ws);
    A.zbuf  = (float*)(ws + 1ull * NELEM * 4);
    A.zq    = (float*)(ws + 2ull * NELEM * 4);
    A.bufA  = (float*)(ws + 3ull * NELEM * 4);
    A.embh  = (_Float16*)(ws + 4ull * NELEM * 4);
    A.wpack = (_Float16*)(ws + 4ull * NELEM * 4 + 131072);
    A.e2    = (float*)   (ws + 4ull * NELEM * 4 + 131072 + 294912);
    A.idxb  = (int*)     (ws + 4ull * NELEM * 4 + 131072 + 294912 + 8192);
    A.part  = (float*)   (ws + 4ull * NELEM * 4 + 131072 + 294912 + 8192 + 4ull * NPIX * 4);

    // ---- per-call prep (deterministic) -------------------------------
    k_zero    <<<NELEM / 256, 256, 0, stream>>>(A.acc);
    k_pack_w  <<<147456 / 256, 256, 0, stream>>>(pw1, pw2, rw1, rw2, A.wpack);
    k_pack_emb<<<65536 / 256, 256, 0, stream>>>(emb, A.embh);
    k_emb_norm<<<VOCAB / 256, 256, 0, stream>>>(emb, A.e2);

    launch_scale<16>(A, 0, stream);
    launch_scale<32>(A, 1, stream);
    launch_scale<48>(A, 2, stream);
    launch_scale<64>(A, 3, stream);

    // outputs: sigmoid(acc) | idx0..idx3 (as float) | loss
    k_sigmoid<<<NELEM / 256, 256, 0, stream>>>(A.acc, out);
    k_idx_out<<<(4 * NPIX) / 256, 256, 0, stream>>>(A.idxb, out + NELEM);
    k_loss_out<<<1, 256, 0, stream>>>(A.part, out + NELEM + 4ull * NPIX);
}
